// SwinTransformer2D_Adapter_AVQA_63299228009336
// MI455X (gfx1250) — matrix operand order repaired and checked by reading, not compiled, not run on hardware
//
#include <hip/hip_runtime.h>

// ---- problem constants -----------------------------------------------------
#define DIM    384
#define HEADS  12
#define N_TOK  49
#define NPAD   64
#define NW     64
#define NBLK   4096     // B_ windows
#define HD     32       // head dim

// ---- WMMA types ------------------------------------------------------------
typedef __attribute__((ext_vector_type(16))) __bf16 v16bf;
typedef __attribute__((ext_vector_type(8)))  float  v8f;

union Frag16 { v16bf v; unsigned short u[16]; uint4 q[2]; };
union Acc8   { v8f   v; float f[8]; };

__device__ __forceinline__ unsigned short f2bf(float f) {
    unsigned int u = __float_as_uint(f);
    return (unsigned short)((u + 0x7FFFu + ((u >> 16) & 1u)) >> 16);
}

// ---- workspace layout ------------------------------------------------------
#define PB_ELEMS (HEADS * NW * NPAD * NPAD)   // 3,145,728 f32  (bias+mask, padded)
#define WQ_ELEMS (72 * 12 * 512)              // 442,368 bf16   (qkv_w packed B-frags)
#define WP_ELEMS (24 * 12 * 512)              // 147,456 bf16   (proj_w packed B-frags)

// ---- prep: pack weights to bf16 B-fragment layout; build bias+mask table ---
__global__ __launch_bounds__(256) void prep_kernel(
    const float* __restrict__ qkv_w,
    const float* __restrict__ proj_w,
    const float* __restrict__ bias_table,   // (169, 12)
    const float* __restrict__ mask,         // (64, 49, 49)
    const int*   __restrict__ rel_index,    // (49, 49)
    float* __restrict__ PB,
    unsigned short* __restrict__ Wq,
    unsigned short* __restrict__ Wp)
{
    long long g = (long long)blockIdx.x * blockDim.x + threadIdx.x;
    if (g < PB_ELEMS) {
        int j = (int)(g & 63), i = (int)((g >> 6) & 63);
        int w = (int)((g >> 12) & 63), h = (int)(g >> 18);
        float v;
        if (i < N_TOK && j < N_TOK) {
            int ridx = rel_index[i * N_TOK + j];
            v = bias_table[ridx * HEADS + h] + mask[(w * N_TOK + i) * N_TOK + j];
        } else if (i < N_TOK) {
            v = -1e30f;        // invalid key column -> exp underflows to 0
        } else {
            v = 0.f;           // padding query row: keep finite (never stored)
        }
        PB[g] = v;
        return;
    }
    g -= PB_ELEMS;
    if (g < WQ_ELEMS) {
        int e = (int)(g & 15), lane = (int)((g >> 4) & 31), t = (int)(g >> 9);
        int ks = t % 12, nt = t / 12;
        int n = lane & 15;
        int k = ks * 32 + (e & 7) + ((e & 8) ? 16 : 0) + ((lane & 16) ? 8 : 0);
        Wq[g] = f2bf(qkv_w[(nt * 16 + n) * DIM + k]);
        return;
    }
    g -= WQ_ELEMS;
    if (g < WP_ELEMS) {
        int e = (int)(g & 15), lane = (int)((g >> 4) & 31), t = (int)(g >> 9);
        int ks = t % 12, nt = t / 12;
        int n = lane & 15;
        int k = ks * 32 + (e & 7) + ((e & 8) ? 16 : 0) + ((lane & 16) ? 8 : 0);
        Wp[g] = f2bf(proj_w[(nt * 16 + n) * DIM + k]);
    }
}

// ---- fused windowed attention: one workgroup (8 wave32) per window ---------
// LDS (ushort units):
//   Xs  [64][392]      @ 0       x bf16, later reused as attn output (pre-proj)
//   Qs  [12][64][40]   @ 25088   scaled q, row-major per head
//   Ks  [12][64][40]   @ 55808   k, row-major per head
//   VTs [12][32][72]   @ 86528   v transposed (dim-major) per head
//   Ps  [8][64][72]    @ 114176  per-wave softmax probabilities
// total 151,040 ushorts = 302,080 B  (< 320 KB WGP LDS)
__global__ __launch_bounds__(256) void swin_attn_kernel(
    const float* __restrict__ x,
    const float* __restrict__ qkv_b,
    const float* __restrict__ proj_b,
    const float* __restrict__ PB,
    const unsigned short* __restrict__ Wq,
    const unsigned short* __restrict__ Wp,
    float* __restrict__ out)
{
    __shared__ __align__(16) unsigned short smem[151040];

    const int b    = blockIdx.x;
    const int w    = b & (NW - 1);
    const int tid  = threadIdx.x;
    const int wave = tid >> 5;
    const int lane = tid & 31;
    const int lhi  = lane >> 4;     // 0 for lanes 0-15, 1 for 16-31
    const int ln   = lane & 15;

    unsigned short* Xs  = smem;
    unsigned short* Qs  = smem + 25088;
    unsigned short* Ks  = smem + 55808;
    unsigned short* VTs = smem + 86528;
    unsigned short* Ps  = smem + 114176 + wave * 4608;

    // ---- stage 1: x -> LDS bf16, rows 49..63 zero --------------------------
    for (int idx = tid; idx < NPAD * DIM; idx += 256) {
        int t = idx / DIM, d = idx - t * DIM;
        float v = (t < N_TOK) ? x[((size_t)b * N_TOK + t) * DIM + d] : 0.f;
        Xs[t * 392 + d] = f2bf(v);
    }
    __syncthreads();

    // ---- stage 2: QKV GEMM (M=64, N=1152, K=384) ---------------------------
    for (int nt = wave; nt < 72; nt += 8) {
        int o0    = nt * 16;
        int part  = o0 / DIM;          // 0=q 1=k 2=v
        int rem   = o0 - part * DIM;
        int head  = rem >> 5;
        int dbase = rem & 31;          // 0 or 16

        Acc8 acc[4];
        for (int mt = 0; mt < 4; ++mt)
            for (int r = 0; r < 8; ++r) acc[mt].f[r] = 0.f;

        for (int ks = 0; ks < 12; ++ks) {
            Frag16 bf;
            const uint4* wp4 = (const uint4*)(Wq + ((size_t)(nt * 12 + ks) * 32 + lane) * 16);
            bf.q[0] = wp4[0]; bf.q[1] = wp4[1];
            for (int mt = 0; mt < 4; ++mt) {
                Frag16 af;
                const unsigned short* ap = Xs + (mt * 16 + ln) * 392 + ks * 32 + lhi * 8;
                af.q[0] = *(const uint4*)(ap);
                af.q[1] = *(const uint4*)(ap + 16);
                acc[mt].v = __builtin_amdgcn_wmma_f32_16x16x32_bf16(
                    false, af.v, false, bf.v, (short)0, acc[mt].v, false, false);
            }
        }

        float bn = qkv_b[o0 + ln];
        for (int mt = 0; mt < 4; ++mt)
            for (int r = 0; r < 8; ++r) {
                int tok = mt * 16 + r + lhi * 8;
                int d   = dbase + ln;
                float v = acc[mt].f[r] + bn;
                if (part == 0)      Qs[head * 2560 + tok * 40 + d] = f2bf(v * 0.1767766953f);
                else if (part == 1) Ks[head * 2560 + tok * 40 + d] = f2bf(v);
                else                VTs[head * 2304 + d * 72 + tok] = f2bf(v);
            }
    }
    __syncthreads();

    // ---- stage 3: per-head attention (heads striped over waves) ------------
    for (int h = wave; h < HEADS; h += 8) {
        const unsigned short* Qh = Qs + h * 2560;
        const unsigned short* Kh = Ks + h * 2560;
        const unsigned short* Vh = VTs + h * 2304;
        const float* pbh = PB + (size_t)(h * NW + w) * NPAD * NPAD;

        // S = q k^T + (bias+mask); softmax row-wise; P -> LDS (bf16)
        for (int mt = 0; mt < 4; ++mt) {
            Frag16 af;
            const unsigned short* ap = Qh + (mt * 16 + ln) * 40 + lhi * 8;
            af.q[0] = *(const uint4*)(ap);
            af.q[1] = *(const uint4*)(ap + 16);

            Acc8 s[4];
            for (int jt = 0; jt < 4; ++jt) {
                for (int r = 0; r < 8; ++r) s[jt].f[r] = 0.f;
                Frag16 bf;
                const unsigned short* kp = Kh + (jt * 16 + ln) * 40 + lhi * 8;
                bf.q[0] = *(const uint4*)(kp);
                bf.q[1] = *(const uint4*)(kp + 16);
                s[jt].v = __builtin_amdgcn_wmma_f32_16x16x32_bf16(
                    false, af.v, false, bf.v, (short)0, s[jt].v, false, false);
            }

            for (int r = 0; r < 8; ++r) {
                int i = mt * 16 + r + lhi * 8;
                float vv[4], mx = -3.0e38f;
                for (int jt = 0; jt < 4; ++jt) {
                    float sc = s[jt].f[r] + pbh[i * 64 + jt * 16 + ln];
                    vv[jt] = sc;
                    mx = fmaxf(mx, sc);
                }
                // reduce within 16-lane group (rows live per half-wave)
                for (int off = 1; off < 16; off <<= 1)
                    mx = fmaxf(mx, __shfl_xor(mx, off, 32));
                float sum = 0.f;
                for (int jt = 0; jt < 4; ++jt) { vv[jt] = __expf(vv[jt] - mx); sum += vv[jt]; }
                for (int off = 1; off < 16; off <<= 1)
                    sum += __shfl_xor(sum, off, 32);
                float inv = 1.0f / sum;
                for (int jt = 0; jt < 4; ++jt)
                    Ps[i * 72 + jt * 16 + ln] = f2bf(vv[jt] * inv);
            }
        }

        // out_h = P @ V  (K = 64 as two K=32 steps), write bf16 into Xs region
        for (int mt = 0; mt < 4; ++mt) {
            for (int dt = 0; dt < 2; ++dt) {
                Acc8 o8;
                for (int r = 0; r < 8; ++r) o8.f[r] = 0.f;
                for (int kc = 0; kc < 2; ++kc) {
                    Frag16 af;
                    const unsigned short* pp = Ps + (mt * 16 + ln) * 72 + kc * 32 + lhi * 8;
                    af.q[0] = *(const uint4*)(pp);
                    af.q[1] = *(const uint4*)(pp + 16);
                    Frag16 bf;
                    const unsigned short* vp = Vh + (dt * 16 + ln) * 72 + kc * 32 + lhi * 8;
                    bf.q[0] = *(const uint4*)(vp);
                    bf.q[1] = *(const uint4*)(vp + 16);
                    o8.v = __builtin_amdgcn_wmma_f32_16x16x32_bf16(
                        false, af.v, false, bf.v, (short)0, o8.v, false, false);
                }
                for (int r = 0; r < 8; ++r) {
                    int tok = mt * 16 + r + lhi * 8;
                    Xs[tok * 392 + h * HD + dt * 16 + ln] = f2bf(o8.f[r]);
                }
            }
        }
    }
    __syncthreads();

    // ---- stage 4: output projection (M=64, N=384, K=384) -------------------
    for (int nt = wave; nt < 24; nt += 8) {
        Acc8 acc[4];
        for (int mt = 0; mt < 4; ++mt)
            for (int r = 0; r < 8; ++r) acc[mt].f[r] = 0.f;

        for (int ks = 0; ks < 12; ++ks) {
            Frag16 bf;
            const uint4* wp4 = (const uint4*)(Wp + ((size_t)(nt * 12 + ks) * 32 + lane) * 16);
            bf.q[0] = wp4[0]; bf.q[1] = wp4[1];
            for (int mt = 0; mt < 4; ++mt) {
                Frag16 af;
                const unsigned short* ap = Xs + (mt * 16 + ln) * 392 + ks * 32 + lhi * 8;
                af.q[0] = *(const uint4*)(ap);
                af.q[1] = *(const uint4*)(ap + 16);
                acc[mt].v = __builtin_amdgcn_wmma_f32_16x16x32_bf16(
                    false, af.v, false, bf.v, (short)0, acc[mt].v, false, false);
            }
        }

        float bn = proj_b[nt * 16 + ln];
        for (int mt = 0; mt < 4; ++mt)
            for (int r = 0; r < 8; ++r) {
                int tok = mt * 16 + r + lhi * 8;
                if (tok < N_TOK)
                    out[((size_t)b * N_TOK + tok) * DIM + nt * 16 + ln] = acc[mt].f[r] + bn;
            }
    }
}

// ---- host glue -------------------------------------------------------------
extern "C" void kernel_launch(void* const* d_in, const int* in_sizes, int n_in,
                              void* d_out, int out_size, void* d_ws, size_t ws_size,
                              hipStream_t stream) {
    const float* x      = (const float*)d_in[0];
    const float* qkv_w  = (const float*)d_in[1];
    const float* qkv_b  = (const float*)d_in[2];
    const float* proj_w = (const float*)d_in[3];
    const float* proj_b = (const float*)d_in[4];
    const float* btab   = (const float*)d_in[5];
    const float* mask   = (const float*)d_in[6];
    const int*   ridx   = (const int*)d_in[7];

    float*          PB = (float*)d_ws;
    unsigned short* Wq = (unsigned short*)(PB + PB_ELEMS);
    unsigned short* Wp = Wq + WQ_ELEMS;

    long long prep_total = (long long)PB_ELEMS + WQ_ELEMS + WP_ELEMS;
    int prep_blocks = (int)((prep_total + 255) / 256);
    prep_kernel<<<prep_blocks, 256, 0, stream>>>(qkv_w, proj_w, btab, mask, ridx, PB, Wq, Wp);

    swin_attn_kernel<<<NBLK, 256, 0, stream>>>(x, qkv_b, proj_b, PB, Wq, Wp, (float*)d_out);
}
